// ASSM_2817498546616
// MI455X (gfx1250) — compile-verified, matrix operand order checked by reading, not run on hardware
//
#include <hip/hip_runtime.h>
#include <cstdint>
#include <cstddef>

// ---------------- CDNA5 WMMA types ----------------
typedef __attribute__((ext_vector_type(16))) __bf16 v16bf;
typedef __attribute__((ext_vector_type(8)))  float  v8f;

#define DEV static __device__ __forceinline__

// ---------------- model dims ----------------
constexpr int Bsz = 4, STXT = 32, SIMG = 224, IMGM = 2048, SP = 7;
constexpr int HID = 768, INTER = 1536, NH = 24, HD = 64, DST = 64;
constexpr int SEQ = STXT + SIMG;                 // 256
constexpr int INPUT_DIM = HID * SP * SP;         // 37632
constexpr int CONV_DIM = INTER + 2 * DST;        // 1664
constexpr int PROJ = INTER + CONV_DIM + NH;      // 3224
constexpr int PROJP = 3232;                      // stored cols (/16)
constexpr int PROJPAD = 3264;                    // B-panel rows (/64)
constexpr int NBLOCKS = 4, OUTD = 32, KC = 4;
constexpr int OUTPAD = 64;                       // head B-panel rows (/64)
constexpr int HW = SP * SP;                      // 49
constexpr float EPS = 1e-5f;
constexpr float PE_COEF = 0.011992631f;          // ln(10000)/768
constexpr int BUFE = 64 * 32;                    // LDS panel elements (4KB)

// ---------------- async global -> LDS copy (CDNA5) ----------------
// GLOBAL_LOAD_ASYNC_TO_LDS_B128: 16B per lane, tracked by ASYNCcnt.
DEV void async_copy16(const __bf16* gsrc, unsigned ldsoff) {
  asm volatile("global_load_async_to_lds_b128 %0, %1, off"
               :: "v"(ldsoff), "v"((uint64_t)(uintptr_t)gsrc)
               : "memory");
}
DEV void wait_async0() { asm volatile("s_wait_asynccnt 0x0" ::: "memory"); }
DEV void wait_async1() { asm volatile("s_wait_asynccnt 0x1" ::: "memory"); }
DEV void wait_async2() { asm volatile("s_wait_asynccnt 0x2" ::: "memory"); }
DEV unsigned lds_off(const void* p) {            // low 32 bits = LDS offset
  return (unsigned)(uintptr_t)p;
}

// ---------------- WMMA fragment helpers ----------------
// 16-bit A fragment layout (ISA 7.12.2): lane l -> row M = l&15; K chunks
// [base..base+7] and [base+16..base+23], base = (l>>4)*8. B uses the same
// layout on the transposed (NxK) weight panel.
DEV v16bf load_frag(const __bf16* rowp, int k) {
  union { uint4 u[2]; v16bf v; } f;
  f.u[0] = *(const uint4*)(rowp + k);        // 16B ds_load
  f.u[1] = *(const uint4*)(rowp + k + 16);   // 16B ds_load
  return f.v;
}
DEV v8f wmma_bf16(v16bf a, v16bf b, v8f c) {
  return __builtin_amdgcn_wmma_f32_16x16x32_bf16(false, a, false, b,
                                                 (short)0, c, false, false);
}

// -------- double-buffered LDS-staged GEMM core: block tile 64(M) x 64(N) ----
// 8 waves = 2(Mpair) x 4(N); each wave computes 2 m-tiles sharing one B frag.
// Per k-step each thread async-copies one 16B A chunk + one 16B B chunk
// (= 2 async instructions per wave), double buffered: wait asynccnt<=2 keeps
// the next stage in flight while this stage feeds the WMMAs.
DEV void gemm_tile_lds2(const __bf16* __restrict__ A, int lda, int mblk,
                        const __bf16* __restrict__ Bt, int ldb, int nblk,
                        int Kd, __bf16* smA, __bf16* smB,
                        v8f& acc0, v8f& acc1) {
  const int tid = threadIdx.x;
  const int lane = tid & 31, w = tid >> 5;
  const int hi = (lane >> 4) << 3;
  const __bf16* aL0 = smA + ((w >> 2) * 16 + (lane & 15)) * 32;   // rows 0..31
  const __bf16* aL1 = aL0 + 32 * 32;                              // rows 32..63
  const __bf16* bL  = smB + ((w & 3) * 16 + (lane & 15)) * 32;
  const int row = tid >> 2, col = (tid & 3) * 8;                  // 64x4 chunks
  __bf16* aS = smA + row * 32 + col;
  __bf16* bS = smB + row * 32 + col;
  const __bf16* aG = A + (size_t)(mblk + row) * lda + col;
  const __bf16* bG = Bt + (size_t)(nblk + row) * ldb + col;
  acc0 = (v8f){}; acc1 = (v8f){};
  // prologue: stages 0 and 1 in flight
  async_copy16(aG, lds_off(aS));
  async_copy16(bG, lds_off(bS));
  if (Kd > 32) {
    async_copy16(aG + 32, lds_off(aS + BUFE));
    async_copy16(bG + 32, lds_off(bS + BUFE));
  }
  int bi = 0;
  for (int k0 = 0; k0 < Kd; k0 += 32, bi ^= 1) {
    if (k0 + 32 < Kd) wait_async2(); else wait_async0();   // stage k0 complete
    __syncthreads();                                       // visible to all waves
    v16bf a0 = load_frag(aL0 + bi * BUFE, hi);
    v16bf a1 = load_frag(aL1 + bi * BUFE, hi);
    v16bf b  = load_frag(bL  + bi * BUFE, hi);
    acc0 = wmma_bf16(a0, b, acc0);
    acc1 = wmma_bf16(a1, b, acc1);
    __syncthreads();                                       // all reads of buf bi done
    if (k0 + 64 < Kd) {                                    // refill vacated buffer
      async_copy16(aG + k0 + 64, lds_off(aS + bi * BUFE));
      async_copy16(bG + k0 + 64, lds_off(bS + bi * BUFE));
    }
  }
}

// conv3d variant: A gathered strided from image_embs f32 (cvt in-register),
// B (weights) double-buffered via async LDS (1 async instr per wave per stage).
DEV void conv_tile_lds2(const float* __restrict__ img, int mblk,
                        const __bf16* __restrict__ Bt, int nblk, __bf16* smB,
                        v8f& acc0, v8f& acc1) {
  const int tid = threadIdx.x;
  const int lane = tid & 31, w = tid >> 5;
  const int hi = (lane >> 4) << 3;
  const int ma = mblk + (w >> 2) * 16 + (lane & 15);
  const int mb = ma + 32;
  const float* abase0 = img + ((size_t)(ma / HW) * IMGM) * HW + (ma % HW);
  const float* abase1 = img + ((size_t)(mb / HW) * IMGM) * HW + (mb % HW);
  const __bf16* bL = smB + ((w & 3) * 16 + (lane & 15)) * 32;
  const int row = tid >> 2, col = (tid & 3) * 8;
  __bf16* bS = smB + row * 32 + col;
  const __bf16* bG = Bt + (size_t)(nblk + row) * IMGM + col;
  acc0 = (v8f){}; acc1 = (v8f){};
  async_copy16(bG, lds_off(bS));
  async_copy16(bG + 32, lds_off(bS + BUFE));
  int bi = 0;
  for (int k0 = 0; k0 < IMGM; k0 += 32, bi ^= 1) {
    if (k0 + 32 < IMGM) wait_async1(); else wait_async0();
    __syncthreads();
    const float* p0 = abase0 + (size_t)(k0 + hi) * HW;
    const float* p1 = abase1 + (size_t)(k0 + hi) * HW;
    v16bf a0, a1;
#pragma unroll
    for (int e = 0; e < 8; ++e) {
      a0[e]     = (__bf16)p0[(size_t)e * HW];
      a0[e + 8] = (__bf16)p0[(size_t)(e + 16) * HW];
      a1[e]     = (__bf16)p1[(size_t)e * HW];
      a1[e + 8] = (__bf16)p1[(size_t)(e + 16) * HW];
    }
    v16bf b = load_frag(bL + bi * BUFE, hi);
    acc0 = wmma_bf16(a0, b, acc0);
    acc1 = wmma_bf16(a1, b, acc1);
    __syncthreads();
    if (k0 + 64 < IMGM) async_copy16(bG + k0 + 64, lds_off(bS + bi * BUFE));
  }
}

DEV void wave_tile(int& m0, int& n0, int& lane) {
  lane = threadIdx.x & 31;
  const int w = threadIdx.x >> 5;
  m0 = blockIdx.y * 64 + (w >> 2) * 16;   // first of two m-tiles (other +32)
  n0 = blockIdx.x * 64 + (w & 3) * 16;
}

// ---------------- conversion kernels ----------------
__global__ void k_convert_f32_bf16(const float* __restrict__ x,
                                   __bf16* __restrict__ y, int n) {
  int i = blockIdx.x * 256 + threadIdx.x;
  if (i < n) y[i] = (__bf16)x[i];
}

// W (Kd x N, f32) -> Bt (Npad x Kd, bf16), optional per-K scale, zero pad.
__global__ void k_convert_t(const float* __restrict__ W,
                            const float* __restrict__ scale,
                            __bf16* __restrict__ Bt, int Kd, int N, int Npad) {
  long long i = (long long)blockIdx.x * 256 + threadIdx.x;
  long long tot = (long long)Npad * Kd;
  if (i >= tot) return;
  int n = (int)(i / Kd), k = (int)(i % Kd);
  float v = 0.f;
  if (n < N) { v = W[(size_t)k * N + n]; if (scale) v *= scale[k]; }
  Bt[i] = (__bf16)v;
}

// cg[n] = sum_k g[k]*W[k,n]; cb[n] = sum_k beta[k]*W[k,n] + bias[n]
__global__ void k_colvec(const float* __restrict__ W, const float* __restrict__ g,
                         const float* __restrict__ beta, const float* __restrict__ bias,
                         float* __restrict__ cg, float* __restrict__ cb,
                         int Kd, int N) {
  __shared__ float r1[256], r2[256];
  int n = blockIdx.x, tid = threadIdx.x;
  float a = 0.f, b = 0.f;
  for (int k = tid; k < Kd; k += 256) {
    float w = W[(size_t)k * N + n];
    a += g[k] * w; b += beta[k] * w;
  }
  r1[tid] = a; r2[tid] = b; __syncthreads();
  for (int s = 128; s > 0; s >>= 1) {
    if (tid < s) { r1[tid] += r1[tid + s]; r2[tid] += r2[tid + s]; }
    __syncthreads();
  }
  if (tid == 0) { cg[n] = r1[0]; cb[n] = r2[0] + bias[n]; }
}

// per-row mean and rsqrt(var+eps) over bf16 rows
__global__ void k_row_stats(const __bf16* __restrict__ X, int Kd,
                            float* __restrict__ mean, float* __restrict__ inv) {
  __shared__ float r1[256], r2[256];
  int row = blockIdx.x, tid = threadIdx.x;
  const __bf16* x = X + (size_t)row * Kd;
  float s = 0.f, q = 0.f;
  for (int k = tid; k < Kd; k += 256) { float v = (float)x[k]; s += v; q += v * v; }
  r1[tid] = s; r2[tid] = q; __syncthreads();
  for (int st = 128; st > 0; st >>= 1) {
    if (tid < st) { r1[tid] += r1[tid + st]; r2[tid] += r2[tid + st]; }
    __syncthreads();
  }
  if (tid == 0) {
    float m = r1[0] / Kd;
    float var = r2[0] / Kd - m * m;
    mean[row] = m; inv[row] = rsqrtf(var + EPS);
  }
}

// ---------------- GEMM kernels (WMMA bf16, f32 accumulate) ----------------
__global__ void k_conv3d_gemm(const float* __restrict__ img,
                              const __bf16* __restrict__ Wc,
                              const float* __restrict__ bias,
                              __bf16* __restrict__ dst) {
  __shared__ __align__(16) __bf16 smB[2 * BUFE];
  int m0, n0, lane; wave_tile(m0, n0, lane);
  v8f acc0, acc1;
  conv_tile_lds2(img, blockIdx.y * 64, Wc, blockIdx.x * 64, smB, acc0, acc1);
  const int n = n0 + (lane & 15);
  const float bv = bias[n];
#pragma unroll
  for (int t2 = 0; t2 < 2; ++t2) {
    v8f acc = t2 ? acc1 : acc0;
    int mb = m0 + t2 * 32;
#pragma unroll
    for (int i = 0; i < 8; ++i) {
      int m = mb + i + ((lane >> 4) << 3);
      int bs = m / HW, hw = m % HW;
      dst[(size_t)bs * INPUT_DIM + (size_t)n * HW + hw] = (__bf16)(acc[i] + bv);
    }
  }
}

// GEMM with LayerNorm-fold epilogue + sinusoidal PE; row m -> b=m/rpb, s=m%rpb.
__global__ void k_gemm_ln(const __bf16* __restrict__ A, const __bf16* __restrict__ Bt,
                          int Kd, int M,
                          const float* __restrict__ mean, const float* __restrict__ inv,
                          const float* __restrict__ cg, const float* __restrict__ cb,
                          float* __restrict__ h, int rpb, int posoff) {
  __shared__ __align__(16) __bf16 smA[2 * BUFE], smB[2 * BUFE];
  int m0, n0, lane; wave_tile(m0, n0, lane);
  v8f acc0, acc1;
  gemm_tile_lds2(A, Kd, blockIdx.y * 64, Bt, Kd, blockIdx.x * 64, Kd,
                 smA, smB, acc0, acc1);
  if (n0 >= HID) return;                       // wave-uniform, after all barriers
  const int n = n0 + (lane & 15);
  const float cgn = cg[n], cbn = cb[n];
  const float div = __expf(-(float)(n & ~1) * PE_COEF);
#pragma unroll
  for (int t2 = 0; t2 < 2; ++t2) {
    v8f acc = t2 ? acc1 : acc0;
    int mb = m0 + t2 * 32;
#pragma unroll
    for (int i = 0; i < 8; ++i) {
      int m = mb + i + ((lane >> 4) << 3);
      int b = m / rpb, s = m % rpb;
      float ang = (float)s * div;
      float pe = (n & 1) ? __cosf(ang) : __sinf(ang);
      float iv = inv[m];
      h[((size_t)b * SEQ + posoff + s) * HID + n] =
          iv * acc[i] - iv * mean[m] * cgn + cbn + pe;
    }
  }
}

__global__ void k_gemm_plain(const __bf16* __restrict__ A, const __bf16* __restrict__ Bt,
                             int Kd, int M, int Nstore, float* __restrict__ C, int ldc) {
  __shared__ __align__(16) __bf16 smA[2 * BUFE], smB[2 * BUFE];
  int m0, n0, lane; wave_tile(m0, n0, lane);
  v8f acc0, acc1;
  gemm_tile_lds2(A, Kd, blockIdx.y * 64, Bt, Kd, blockIdx.x * 64, Kd,
                 smA, smB, acc0, acc1);
  if (n0 >= Nstore) return;
  const int n = n0 + (lane & 15);
#pragma unroll
  for (int t2 = 0; t2 < 2; ++t2) {
    v8f acc = t2 ? acc1 : acc0;
    int mb = m0 + t2 * 32;
#pragma unroll
    for (int i = 0; i < 8; ++i) {
      int m = mb + i + ((lane >> 4) << 3);
      C[(size_t)m * ldc + n] = acc[i];
    }
  }
}

__global__ void k_gemm_add(const __bf16* __restrict__ A, const __bf16* __restrict__ Bt,
                           int Kd, int M, float* __restrict__ h) {
  __shared__ __align__(16) __bf16 smA[2 * BUFE], smB[2 * BUFE];
  int m0, n0, lane; wave_tile(m0, n0, lane);
  v8f acc0, acc1;
  gemm_tile_lds2(A, Kd, blockIdx.y * 64, Bt, Kd, blockIdx.x * 64, Kd,
                 smA, smB, acc0, acc1);
  if (n0 >= HID) return;
  const int n = n0 + (lane & 15);
#pragma unroll
  for (int t2 = 0; t2 < 2; ++t2) {
    v8f acc = t2 ? acc1 : acc0;
    int mb = m0 + t2 * 32;
#pragma unroll
    for (int i = 0; i < 8; ++i) {
      int m = mb + i + ((lane >> 4) << 3);
      h[(size_t)m * HID + n] += acc[i];
    }
  }
}

__global__ void k_gemm_head(const __bf16* __restrict__ A, const __bf16* __restrict__ Bt,
                            int Kd, int M, const float* __restrict__ hb,
                            float* __restrict__ out) {
  __shared__ __align__(16) __bf16 smA[2 * BUFE], smB[2 * BUFE];
  int m0, n0, lane; wave_tile(m0, n0, lane);
  v8f acc0, acc1;
  gemm_tile_lds2(A, Kd, blockIdx.y * 64, Bt, Kd, blockIdx.x * 64, Kd,
                 smA, smB, acc0, acc1);
  if (n0 >= OUTD) return;
  const int n = n0 + (lane & 15);
  const float bv = hb[n];
#pragma unroll
  for (int t2 = 0; t2 < 2; ++t2) {
    v8f acc = t2 ? acc1 : acc0;
    int mb = m0 + t2 * 32;
#pragma unroll
    for (int i = 0; i < 8; ++i) {
      int m = mb + i + ((lane >> 4) << 3);
      int b = m >> 8, t = m & 255;
      if (t >= STXT)
        out[((size_t)b * SIMG + (t - STXT)) * OUTD + n] = acc[i] + bv;
    }
  }
}

// ---------------- elementwise / norm / scan kernels ----------------
__global__ void k_rmsnorm_bf(const float* __restrict__ h, const float* __restrict__ w,
                             const unsigned char* __restrict__ mask,
                             __bf16* __restrict__ out, int width) {
  __shared__ float red[256];
  int row = blockIdx.x, tid = threadIdx.x;
  const float* x = h + (size_t)row * width;
  float vals[6]; int c = 0; float q = 0.f;
  for (int j = tid; j < width; j += 256) { float v = x[j]; vals[c++] = v; q += v * v; }
  red[tid] = q; __syncthreads();
  for (int s = 128; s > 0; s >>= 1) { if (tid < s) red[tid] += red[tid + s]; __syncthreads(); }
  float scale = rsqrtf(red[0] / width + EPS);
  float mf = mask ? (mask[row] ? 1.f : 0.f) : 1.f;
  c = 0;
  __bf16* o = out + (size_t)row * width;
  for (int j = tid; j < width; j += 256) o[j] = (__bf16)(vals[c++] * scale * w[j] * mf);
}

__global__ void k_conv_silu(const float* __restrict__ proj, const float* __restrict__ cw,
                            const float* __restrict__ cb,
                            const unsigned char* __restrict__ mask,
                            float* __restrict__ xbc) {
  int tid = blockIdx.x * 256 + threadIdx.x;
  if (tid >= Bsz * CONV_DIM) return;
  int b = tid / CONV_DIM, c = tid % CONV_DIM;
  float w0 = cw[c * KC + 0], w1 = cw[c * KC + 1], w2 = cw[c * KC + 2], w3 = cw[c * KC + 3];
  float bias = cb[c];
  float x0 = 0.f, x1 = 0.f, x2 = 0.f;
  for (int t = 0; t < SEQ; ++t) {
    size_t r = (size_t)b * SEQ + t;
    float x3 = proj[r * PROJP + INTER + c];
    float v = bias + w0 * x0 + w1 * x1 + w2 * x2 + w3 * x3;
    v = v / (1.f + __expf(-v));                       // silu
    float mf = mask[b * SEQ + t] ? 1.f : 0.f;
    xbc[r * CONV_DIM + c] = v * mf;
    x0 = x1; x1 = x2; x2 = x3;
  }
}

__global__ void k_dt(const float* __restrict__ proj, const float* __restrict__ dtb,
                     const float* __restrict__ alog,
                     float* __restrict__ dt, float* __restrict__ dA) {
  int i = blockIdx.x * 256 + threadIdx.x;
  if (i >= Bsz * SEQ * NH) return;
  int r = i / NH, hh = i % NH;
  float x = proj[(size_t)r * PROJP + INTER + CONV_DIM + hh] + dtb[hh];
  float sp = (x > 20.f) ? x : log1pf(__expf(x));      // softplus
  dt[i] = sp;
  dA[i] = __expf(sp * (-__expf(alog[hh])));
}

// One 64-thread block per (batch, head): thread p owns state[p][0..63] in regs.
__global__ void k_scan(const float* __restrict__ xbc, const float* __restrict__ dt,
                       const float* __restrict__ dA, const float* __restrict__ Dpar,
                       float* __restrict__ ybuf) {
  __shared__ float Bs[DST], Cs[DST];
  int bh = blockIdx.x;
  int b = bh / NH, hh = bh % NH;
  int p = threadIdx.x;                                // 0..63
  float Dv = Dpar[hh];
  float state[DST];
#pragma unroll
  for (int n = 0; n < DST; ++n) state[n] = 0.f;
  for (int t = 0; t < SEQ; ++t) {
    size_t r = (size_t)b * SEQ + t;
    const float* xr = xbc + r * CONV_DIM;
    Bs[p] = xr[INTER + p];
    Cs[p] = xr[INTER + DST + p];
    __syncthreads();
    float xp = xr[hh * HD + p];
    float dtt = dt[r * NH + hh];
    float dAt = dA[r * NH + hh];
    float coef = dtt * xp;
    float y = 0.f;
#pragma unroll
    for (int n = 0; n < DST; ++n) {
      float st = state[n] * dAt + coef * Bs[n];
      state[n] = st;
      y += st * Cs[n];
    }
    ybuf[r * INTER + hh * HD + p] = y + Dv * xp;
    __syncthreads();
  }
}

__global__ void k_gate_rms(const float* __restrict__ y, const float* __restrict__ proj,
                           const float* __restrict__ gw, __bf16* __restrict__ out) {
  __shared__ float red[256];
  int row = blockIdx.x, tid = threadIdx.x;
  const float* yr = y + (size_t)row * INTER;
  const float* pr = proj + (size_t)row * PROJP;
  float vals[INTER / 256]; int c = 0; float q = 0.f;
  for (int j = tid; j < INTER; j += 256) {
    float z = pr[j];
    float g = z / (1.f + __expf(-z));                 // silu(z)
    float v = yr[j] * g;
    vals[c++] = v; q += v * v;
  }
  red[tid] = q; __syncthreads();
  for (int s = 128; s > 0; s >>= 1) { if (tid < s) red[tid] += red[tid + s]; __syncthreads(); }
  float scale = rsqrtf(red[0] / INTER + EPS);
  c = 0;
  __bf16* o = out + (size_t)row * INTER;
  for (int j = tid; j < INTER; j += 256) o[j] = (__bf16)(vals[c++] * scale * gw[j]);
}

// ---------------- host launch ----------------
extern "C" void kernel_launch(void* const* d_in, const int* in_sizes, int n_in,
                              void* d_out, int out_size, void* d_ws, size_t ws_size,
                              hipStream_t stream) {
  (void)in_sizes; (void)n_in; (void)out_size; (void)ws_size;
  const float* image_embs = (const float*)d_in[0];
  const float* instr      = (const float*)d_in[1];
  const unsigned char* mask = (const unsigned char*)d_in[2];
  const float* conv3d_w   = (const float*)d_in[3];
  const float* conv3d_b   = (const float*)d_in[4];
  const float* ln_img_g   = (const float*)d_in[5];
  const float* ln_img_b   = (const float*)d_in[6];
  const float* ln_ins_g   = (const float*)d_in[7];
  const float* ln_ins_b   = (const float*)d_in[8];
  const float* ins_w      = (const float*)d_in[9];
  const float* ins_b      = (const float*)d_in[10];
  const float* img_w      = (const float*)d_in[11];
  const float* img_b      = (const float*)d_in[12];
  const float* head_w     = (const float*)d_in[13];
  const float* head_b     = (const float*)d_in[14];
  const float* in_proj_w  = (const float*)d_in[15];
  const float* norm_w     = (const float*)d_in[16];
  const float* conv_w     = (const float*)d_in[17];
  const float* conv_b     = (const float*)d_in[18];
  const float* dt_bias    = (const float*)d_in[19];
  const float* A_log      = (const float*)d_in[20];
  const float* Dp         = (const float*)d_in[21];
  const float* gnorm_w    = (const float*)d_in[22];
  const float* out_proj_w = (const float*)d_in[23];
  const float* normf_w    = (const float*)d_in[24];

  char* p = (char*)d_ws;
  auto bump = [&](size_t bytes) -> void* {
    void* r = (void*)p; p += (bytes + 255) & ~(size_t)255; return r;
  };
  // workspace (~171 MB total)
  __bf16* wc_bf   = (__bf16*)bump((size_t)HID * IMGM * 2);
  __bf16* imgs_bf = (__bf16*)bump((size_t)Bsz * SIMG * INPUT_DIM * 2);
  float*  rowm_i  = (float*)bump((size_t)Bsz * SIMG * 4);
  float*  rowi_i  = (float*)bump((size_t)Bsz * SIMG * 4);
  __bf16* wimg_t  = (__bf16*)bump((size_t)HID * INPUT_DIM * 2);
  float*  cg_i    = (float*)bump(HID * 4);
  float*  cb_i    = (float*)bump(HID * 4);
  __bf16* ins_bf  = (__bf16*)bump((size_t)Bsz * STXT * HID * 2);
  float*  rowm_t  = (float*)bump((size_t)Bsz * STXT * 4);
  float*  rowi_t  = (float*)bump((size_t)Bsz * STXT * 4);
  __bf16* wins_t  = (__bf16*)bump((size_t)HID * HID * 2);
  float*  cg_t    = (float*)bump(HID * 4);
  float*  cb_t    = (float*)bump(HID * 4);
  float*  h       = (float*)bump((size_t)Bsz * SEQ * HID * 4);
  __bf16* hn_bf   = (__bf16*)bump((size_t)Bsz * SEQ * HID * 2);
  __bf16* wproj_t = (__bf16*)bump((size_t)PROJPAD * HID * 2);
  float*  proj    = (float*)bump((size_t)Bsz * SEQ * PROJP * 4);
  float*  xbc     = (float*)bump((size_t)Bsz * SEQ * CONV_DIM * 4);
  float*  dtv     = (float*)bump((size_t)Bsz * SEQ * NH * 4);
  float*  dAv     = (float*)bump((size_t)Bsz * SEQ * NH * 4);
  float*  ybuf    = (float*)bump((size_t)Bsz * SEQ * INTER * 4);
  __bf16* ybf     = (__bf16*)bump((size_t)Bsz * SEQ * INTER * 2);
  __bf16* wout_t  = (__bf16*)bump((size_t)HID * INTER * 2);
  __bf16* whead_t = (__bf16*)bump((size_t)OUTPAD * HID * 2);

  const dim3 blk(256);
  const int Mrows = Bsz * SEQ;                         // 1024

  // ---- image path: conv3d GEMM -> (LN folded) projection ----
  { int n = HID * IMGM;
    k_convert_f32_bf16<<<(n + 255) / 256, blk, 0, stream>>>(conv3d_w, wc_bf, n); }
  { dim3 g(HID / 64, (Bsz * SIMG * HW) / 64);
    k_conv3d_gemm<<<g, blk, 0, stream>>>(image_embs, wc_bf, conv3d_b, imgs_bf); }
  k_row_stats<<<Bsz * SIMG, blk, 0, stream>>>(imgs_bf, INPUT_DIM, rowm_i, rowi_i);
  { long long tot = (long long)HID * INPUT_DIM;
    k_convert_t<<<(unsigned)((tot + 255) / 256), blk, 0, stream>>>(
        img_w, ln_img_g, wimg_t, INPUT_DIM, HID, HID); }
  k_colvec<<<HID, blk, 0, stream>>>(img_w, ln_img_g, ln_img_b, img_b, cg_i, cb_i,
                                    INPUT_DIM, HID);
  { dim3 g(HID / 64, (Bsz * SIMG) / 64);
    k_gemm_ln<<<g, blk, 0, stream>>>(imgs_bf, wimg_t, INPUT_DIM, Bsz * SIMG,
                                     rowm_i, rowi_i, cg_i, cb_i, h, SIMG, STXT); }

  // ---- instruction path ----
  { int n = Bsz * STXT * HID;
    k_convert_f32_bf16<<<(n + 255) / 256, blk, 0, stream>>>(instr, ins_bf, n); }
  k_row_stats<<<Bsz * STXT, blk, 0, stream>>>(ins_bf, HID, rowm_t, rowi_t);
  { long long tot = (long long)HID * HID;
    k_convert_t<<<(unsigned)((tot + 255) / 256), blk, 0, stream>>>(
        ins_w, ln_ins_g, wins_t, HID, HID, HID); }
  k_colvec<<<HID, blk, 0, stream>>>(ins_w, ln_ins_g, ln_ins_b, ins_b, cg_t, cb_t,
                                    HID, HID);
  { dim3 g(HID / 64, (Bsz * STXT) / 64);
    k_gemm_ln<<<g, blk, 0, stream>>>(ins_bf, wins_t, HID, Bsz * STXT,
                                     rowm_t, rowi_t, cg_t, cb_t, h, STXT, 0); }

  // ---- Mamba2 blocks ----
  for (int l = 0; l < NBLOCKS; ++l) {
    k_rmsnorm_bf<<<Mrows, blk, 0, stream>>>(h, norm_w + (size_t)l * HID, mask,
                                            hn_bf, HID);
    { long long tot = (long long)PROJPAD * HID;
      k_convert_t<<<(unsigned)((tot + 255) / 256), blk, 0, stream>>>(
          in_proj_w + (size_t)l * HID * PROJ, nullptr, wproj_t, HID, PROJ, PROJPAD); }
    { dim3 g(PROJPAD / 64, Mrows / 64);
      k_gemm_plain<<<g, blk, 0, stream>>>(hn_bf, wproj_t, HID, Mrows, PROJP,
                                          proj, PROJP); }
    { int n = Bsz * CONV_DIM;
      k_conv_silu<<<(n + 255) / 256, blk, 0, stream>>>(
          proj, conv_w + (size_t)l * CONV_DIM * KC, conv_b + (size_t)l * CONV_DIM,
          mask, xbc); }
    { int n = Mrows * NH;
      k_dt<<<(n + 255) / 256, blk, 0, stream>>>(proj, dt_bias + l * NH,
                                                A_log + l * NH, dtv, dAv); }
    k_scan<<<Bsz * NH, dim3(64), 0, stream>>>(xbc, dtv, dAv, Dp + l * NH, ybuf);
    k_gate_rms<<<Mrows, blk, 0, stream>>>(ybuf, proj, gnorm_w + (size_t)l * INTER, ybf);
    { long long tot = (long long)HID * INTER;
      k_convert_t<<<(unsigned)((tot + 255) / 256), blk, 0, stream>>>(
          out_proj_w + (size_t)l * INTER * HID, nullptr, wout_t, INTER, HID, HID); }
    { dim3 g(HID / 64, Mrows / 64);
      k_gemm_add<<<g, blk, 0, stream>>>(ybf, wout_t, INTER, Mrows, h); }
  }

  // ---- final norm + head ----
  k_rmsnorm_bf<<<Mrows, blk, 0, stream>>>(h, normf_w, nullptr, hn_bf, HID);
  { long long tot = (long long)OUTPAD * HID;
    k_convert_t<<<(unsigned)((tot + 255) / 256), blk, 0, stream>>>(
        head_w, nullptr, whead_t, HID, OUTD, OUTPAD); }
  { dim3 g(OUTPAD / 64, Mrows / 64);
    k_gemm_head<<<g, blk, 0, stream>>>(hn_bf, whead_t, HID, Mrows, head_b,
                                       (float*)d_out); }
}
